// Sg2ImModel_16037407883973
// MI455X (gfx1250) — compile-verified, matrix operand order checked by reading, not run on hardware
//
#include <hip/hip_runtime.h>

// ---------------- CDNA5 WMMA types ----------------
typedef __attribute__((ext_vector_type(16))) __bf16 v16bf;
typedef __attribute__((ext_vector_type(8)))  float  v8f;

#define O_NODES 100000
#define T_EDGES 150000
#define HID 512
#define DD  128

// GEMM tiling: 128x128x32, 256 threads (8 wave32), each wave -> 32x64 of C
#define BM 128
#define BN 128
#define BK 32
#define PAD 8

enum { A_BF16 = 0, A_F32 = 1, A_GCONV = 2, A_REL = 3 };
enum { E_BF16 = 0, E_SCATTER = 1, E_F32 = 2 };

struct GArgs {
  int M, N, K;
  const __bf16* Wt;    // weights bf16, TRANSPOSED [N][K] (pre-converted)
  const float*  bias;  // [N] f32
  // A sources (mode-dependent)
  const __bf16* Abf;  int lda;
  const float*  Af32;
  const __bf16* objIn; const __bf16* predIn; int din;   // A_GCONV
  const __bf16* boxes; const __bf16* orig;              // A_REL
  const int* sArr; const int* oArr;
  // epilogues
  __bf16* Cbf;                       // E_BF16
  float*  Cf32; __bf16* Cbf2;        // E_F32 (+optional bf16 copy)
  float*  pooled; __bf16* predOut;   // E_SCATTER (N == 1152)
};

union Frag { unsigned int u[8]; v16bf v; };

// K index inside a 16-bit 16x32 WMMA fragment (ISA 7.12.2), per VGPR pair v and lane-half
__device__ __forceinline__ int frag_k(int v, int half) {
  return (v < 4) ? (2 * v + 8 * half) : (16 + 2 * (v - 4) + 8 * half);
}

// CDNA5 async global->LDS copy, 16 bytes per lane (VGLOBAL GLOBAL_LOAD_ASYNC_TO_LDS_B128).
// ldsOff: LDS byte offset (generic LDS pointers keep the offset in addr[31:0], ISA 10.2).
__device__ __forceinline__ void async_copy16(unsigned ldsOff, const void* gptr) {
  asm volatile("global_load_async_to_lds_b128 %0, %1, off"
               :: "v"(ldsOff), "v"((unsigned long long)(uintptr_t)gptr)
               : "memory");
}
__device__ __forceinline__ void wait_async0() {
  asm volatile("s_wait_asynccnt 0" ::: "memory");
}
__device__ __forceinline__ void lds_zero16(void* p) {
  unsigned long long* z = (unsigned long long*)p;
  z[0] = 0ull; z[1] = 0ull;
}

// Fill one k-step's A/B tiles (buffer-selected by caller).
template <int AM>
__device__ __forceinline__ void fill_tiles(const GArgs& g,
                                           __bf16 (*tA)[BK + PAD],
                                           __bf16 (*tB)[BK + PAD],
                                           int bM, int bN, int k0, int tid) {
  // ---- A tile: 128x32 bf16; each thread owns one row-half (two 16B chunks) ----
  const int r  = tid >> 1;
  const int ks = (tid & 1) * 16;     // element offset of this thread's 16-elem span
  const int gr = bM + r;
  if constexpr (AM == A_BF16 || AM == A_GCONV) {
    if (gr < g.M) {
      if constexpr (AM == A_BF16) {
        const __bf16* src = g.Abf + (long long)gr * g.lda + (k0 + ks);
        async_copy16((unsigned)(uintptr_t)&tA[r][ks], src);
        async_copy16((unsigned)(uintptr_t)&tA[r][ks + 8], src + 8);
      } else { // A_GCONV: cur_t = [obj[s] | pred | obj[o]]; chunk never crosses a region
        const int d = g.din;
        const int sI = g.sArr[gr], oI = g.oArr[gr];
#pragma unroll
        for (int c = 0; c < 2; ++c) {
          const int kc = ks + c * 8;
          const int gk = k0 + kc;
          const __bf16* src;
          if (gk < d)          src = g.objIn  + (long long)sI * d + gk;
          else if (gk < 2 * d) src = g.predIn + (long long)gr * d + (gk - d);
          else                 src = g.objIn  + (long long)oI * d + (gk - 2 * d);
          async_copy16((unsigned)(uintptr_t)&tA[r][kc], src);
        }
      }
    } else {
      lds_zero16(&tA[r][ks]);
      lds_zero16(&tA[r][ks + 8]);
    }
  } else if constexpr (AM == A_F32) {
    // f32 -> bf16 convert path (pooled): 16 contiguous floats per thread
#pragma unroll
    for (int j = 0; j < 16; ++j) {
      const int gk = k0 + ks + j;
      __bf16 v = (__bf16)0.0f;
      if (gr < g.M && gk < g.K) v = (__bf16)g.Af32[(long long)gr * g.lda + gk];
      tA[r][ks + j] = v;
    }
  } else { // A_REL: [boxes[s](4) | boxes[o](4) | orig[s](64) | orig[o](64)], K=136
    const int sI = (gr < g.M) ? g.sArr[gr] : 0;
    const int oI = (gr < g.M) ? g.oArr[gr] : 0;
#pragma unroll
    for (int j = 0; j < 16; ++j) {
      const int gk = k0 + ks + j;
      __bf16 val = (__bf16)0.0f;
      if (gr < g.M && gk < g.K) {
        if (gk < 4)        val = g.boxes[sI * 4 + gk];
        else if (gk < 8)   val = g.boxes[oI * 4 + (gk - 4)];
        else if (gk < 72)  val = g.orig[(long long)sI * 64 + (gk - 8)];
        else               val = g.orig[(long long)oI * 64 + (gk - 72)];
      }
      tA[r][ks + j] = val;
    }
  }

  // ---- B tile: Wt is [N][K] bf16, async copy straight into transposed LDS layout ----
  {
    const int nr = tid >> 1;
    const int gn = bN + nr;
#pragma unroll
    for (int c = 0; c < 2; ++c) {
      const int kc = ks + c * 8;
      const int gk = k0 + kc;
      if (gn < g.N && gk < g.K) {  // K%8==0 for all GEMMs -> chunk fully valid
        async_copy16((unsigned)(uintptr_t)&tB[nr][kc],
                     g.Wt + (long long)gn * g.K + gk);
      } else {
        lds_zero16(&tB[nr][kc]);
      }
    }
  }
}

template <int AM, int EM>
__global__ __launch_bounds__(256) void gemm_wmma(GArgs g) {
  // double-buffered tiles: DMA for step kt+1 overlaps WMMA on step kt
  __shared__ __bf16 sA[2][BM][BK + PAD];   // row stride 80B, 16B aligned chunks
  __shared__ __bf16 sB[2][BN][BK + PAD];   // transposed: [n][k]

  const int tid  = threadIdx.x;
  const int bM   = blockIdx.x * BM;
  const int bN   = blockIdx.y * BN;
  const int wave = tid >> 5;
  const int lane = tid & 31;
  const int wm   = (wave & 3) * 32;   // 4 wave-rows
  const int wn   = (wave >> 2) * 64;  // 2 wave-cols
  const int lrow = lane & 15;
  const int half = lane >> 4;

  v8f acc[2][4];
#pragma unroll
  for (int i = 0; i < 2; ++i)
#pragma unroll
    for (int j = 0; j < 4; ++j)
#pragma unroll
      for (int e = 0; e < 8; ++e) acc[i][j][e] = 0.0f;

  const int numKT = (g.K + BK - 1) / BK;

  fill_tiles<AM>(g, sA[0], sB[0], bM, bN, 0, tid);

  for (int kt = 0; kt < numKT; ++kt) {
    const int cur = kt & 1;
    wait_async0();        // this wave's DMA into buf[cur] done
    __syncthreads();      // all waves' DMA visible; all prior readers of buf[cur^1] done
    if (kt + 1 < numKT)
      fill_tiles<AM>(g, sA[cur ^ 1], sB[cur ^ 1], bM, bN, (kt + 1) * BK, tid);

    // ---- fragments from LDS (ISA 7.12.2 16-bit layouts; B symmetric via transposed tile) ----
    Frag aF[2], bF[4];
#pragma unroll
    for (int i = 0; i < 2; ++i) {
      const int r = wm + i * 16 + lrow;
#pragma unroll
      for (int v = 0; v < 8; ++v)
        aF[i].u[v] = *(const unsigned int*)&sA[cur][r][frag_k(v, half)];
    }
#pragma unroll
    for (int jj = 0; jj < 4; ++jj) {
      const int c = wn + jj * 16 + lrow;
#pragma unroll
      for (int v = 0; v < 8; ++v)
        bF[jj].u[v] = *(const unsigned int*)&sB[cur][c][frag_k(v, half)];
    }

#pragma unroll
    for (int i = 0; i < 2; ++i)
#pragma unroll
      for (int jj = 0; jj < 4; ++jj)
        acc[i][jj] = __builtin_amdgcn_wmma_f32_16x16x32_bf16(
            false, aF[i].v, false, bF[jj].v, (short)0, acc[i][jj], false, false);
  }

  // ---- epilogue: bias + relu, per-mode store (C/D layout: row = r + 8*half, col = lane&15) ----
  float bb[4];
#pragma unroll
  for (int jj = 0; jj < 4; ++jj) {
    const int gCol = bN + wn + jj * 16 + lrow;
    bb[jj] = (gCol < g.N) ? g.bias[gCol] : 0.0f;
  }
#pragma unroll
  for (int i = 0; i < 2; ++i) {
#pragma unroll
    for (int jj = 0; jj < 4; ++jj) {
#pragma unroll
      for (int r = 0; r < 8; ++r) {
        const int gRow = bM + wm + i * 16 + r + half * 8;
        const int gCol = bN + wn + jj * 16 + lrow;
        if (gRow < g.M && gCol < g.N) {
          float v = fmaxf(acc[i][jj][r] + bb[jj], 0.0f);
          if constexpr (EM == E_BF16) {
            g.Cbf[(long long)gRow * g.N + gCol] = (__bf16)v;
          } else if constexpr (EM == E_F32) {
            g.Cf32[(long long)gRow * g.N + gCol] = v;
            if (g.Cbf2) g.Cbf2[(long long)gRow * g.N + gCol] = (__bf16)v;
          } else { // E_SCATTER: new_t split -> segment-sum into pooled + pred slice
            if (gCol < HID) {
              atomicAdd(&g.pooled[(long long)g.sArr[gRow] * HID + gCol], v);
            } else if (gCol < HID + DD) {
              g.predOut[(long long)gRow * DD + (gCol - HID)] = (__bf16)v;
            } else {
              atomicAdd(&g.pooled[(long long)g.oArr[gRow] * HID + (gCol - HID - DD)], v);
            }
          }
        }
      }
    }
  }
}

// ---------------- small helper kernels ----------------
__global__ void k_extract(const int* __restrict__ t, int* s, int* o, int n) {
  int i = blockIdx.x * blockDim.x + threadIdx.x;
  if (i < n) { s[i] = t[3 * i]; o[i] = t[3 * i + 2]; }
}
__global__ void k_zero(float* p, long long n) {
  long long i = (long long)blockIdx.x * blockDim.x + threadIdx.x;
  if (i < n) p[i] = 0.0f;
}
__global__ void k_counts(const int* s, const int* o, float* c, int n) {
  int i = blockIdx.x * blockDim.x + threadIdx.x;
  if (i < n) { atomicAdd(&c[s[i]], 1.0f); atomicAdd(&c[o[i]], 1.0f); }
}
__global__ void k_embed_obj(const int* idx, const float* emb, __bf16* out, long long n) {
  long long i = (long long)blockIdx.x * blockDim.x + threadIdx.x;
  if (i < n) { int r = (int)(i >> 6), k = (int)(i & 63);
    out[i] = (__bf16)emb[(long long)idx[r] * 64 + k]; }
}
__global__ void k_embed_pred(const int* trip, const float* emb, __bf16* out, long long n) {
  long long i = (long long)blockIdx.x * blockDim.x + threadIdx.x;
  if (i < n) { int e = (int)(i >> 6), k = (int)(i & 63);
    out[i] = (__bf16)emb[(long long)trip[3 * e + 1] * 64 + k]; }
}
__global__ void k_divide(float* pooled, const float* cnt, long long n) {
  long long i = (long long)blockIdx.x * blockDim.x + threadIdx.x;
  if (i < n) pooled[i] /= fmaxf(cnt[i >> 9], 1.0f);  // 512 cols per node
}
// convert weight f32 [K][N] -> bf16 transposed [N][K]
__global__ void k_w2bfT(const float* __restrict__ W, __bf16* __restrict__ Wt, int K, int N) {
  long long i = (long long)blockIdx.x * blockDim.x + threadIdx.x;
  if (i < (long long)K * N) {
    int k = (int)(i / N), n = (int)(i % N);
    Wt[(long long)n * K + k] = (__bf16)W[i];
  }
}

// ---------------- host orchestration ----------------
static inline dim3 ggrid(int M, int N) { return dim3((M + BM - 1) / BM, (N + BN - 1) / BN); }

extern "C" void kernel_launch(void* const* d_in, const int* in_sizes, int n_in,
                              void* d_out, int out_size, void* d_ws, size_t ws_size,
                              hipStream_t stream) {
  (void)in_sizes; (void)n_in; (void)out_size; (void)ws_size;
  // pytree order: objs | box_net(b,w,b,w) | 5x gconv(net1 b,w,b,w ; net2 b,w,b,w) |
  //               obj_emb | pred_emb | rel_aux(b,w,b,w) | triples
  const int*   objs     = (const int*)d_in[0];
  const float* box_b0   = (const float*)d_in[1];
  const float* box_w0   = (const float*)d_in[2];
  const float* box_b1   = (const float*)d_in[3];
  const float* box_w1   = (const float*)d_in[4];
  const float* obj_emb  = (const float*)d_in[45];
  const float* pred_emb = (const float*)d_in[46];
  const float* rel_b0   = (const float*)d_in[47];
  const float* rel_w0   = (const float*)d_in[48];
  const float* rel_b1   = (const float*)d_in[49];
  const float* rel_w1   = (const float*)d_in[50];
  const int*   triples  = (const int*)d_in[51];

  char*  ws  = (char*)d_ws;
  size_t off = 0;
  auto alloc = [&](size_t bytes) -> char* {
    char* p = ws + off; off += (bytes + 255) & ~(size_t)255; return p;
  };
  int*    sArr    = (int*)alloc((size_t)T_EDGES * 4);
  int*    oArr    = (int*)alloc((size_t)T_EDGES * 4);
  float*  counts  = (float*)alloc((size_t)O_NODES * 4);
  __bf16* objOrig = (__bf16*)alloc((size_t)O_NODES * 64 * 2);
  __bf16* pred0   = (__bf16*)alloc((size_t)T_EDGES * 64 * 2);
  __bf16* objP[2] = {(__bf16*)alloc((size_t)O_NODES * 128 * 2),
                     (__bf16*)alloc((size_t)O_NODES * 128 * 2)};
  __bf16* predP[2]= {(__bf16*)alloc((size_t)T_EDGES * 128 * 2),
                     (__bf16*)alloc((size_t)T_EDGES * 128 * 2)};
  __bf16* h1      = (__bf16*)alloc((size_t)T_EDGES * HID * 2);
  __bf16* h2      = (__bf16*)alloc((size_t)O_NODES * HID * 2);
  float*  pooled  = (float*)alloc((size_t)O_NODES * HID * 4);
  __bf16* boxesBf = (__bf16*)alloc((size_t)O_NODES * 4 * 2);

  // bf16 transposed weight buffers + conversion (once per launch; deterministic)
  auto conv = [&](const float* W, int K, int N) -> __bf16* {
    __bf16* Wt = (__bf16*)alloc((size_t)K * N * 2);
    long long n = (long long)K * N;
    k_w2bfT<<<(unsigned)((n + 255) / 256), 256, 0, stream>>>(W, Wt, K, N);
    return Wt;
  };

  // setup (re-done every call: deterministic under graph replay)
  k_extract<<<(T_EDGES + 255) / 256, 256, 0, stream>>>(triples, sArr, oArr, T_EDGES);
  k_zero<<<(O_NODES + 255) / 256, 256, 0, stream>>>(counts, O_NODES);
  k_counts<<<(T_EDGES + 255) / 256, 256, 0, stream>>>(sArr, oArr, counts, T_EDGES);
  { long long n = (long long)O_NODES * 64;
    k_embed_obj<<<(unsigned)((n + 255) / 256), 256, 0, stream>>>(objs, obj_emb, objOrig, n); }
  { long long n = (long long)T_EDGES * 64;
    k_embed_pred<<<(unsigned)((n + 255) / 256), 256, 0, stream>>>(triples, pred_emb, pred0, n); }

  __bf16* wtG[5][4];
  for (int l = 0; l < 5; ++l) {
    const int din = (l == 0) ? 64 : 128;
    const int gi = 5 + l * 8;
    wtG[l][0] = conv((const float*)d_in[gi + 1], 3 * din, HID);       // W1a
    wtG[l][1] = conv((const float*)d_in[gi + 3], HID, 2 * HID + DD);  // W1b
    wtG[l][2] = conv((const float*)d_in[gi + 5], HID, HID);           // W2a
    wtG[l][3] = conv((const float*)d_in[gi + 7], HID, DD);            // W2b
  }
  __bf16* wtBox0 = conv(box_w0, DD, HID);
  __bf16* wtBox1 = conv(box_w1, HID, 4);
  __bf16* wtRel0 = conv(rel_w0, 136, HID);
  __bf16* wtRel1 = conv(rel_w1, HID, 46);

  // 5 gconv layers
  for (int l = 0; l < 5; ++l) {
    const int din = (l == 0) ? 64 : 128;
    const __bf16* oin  = (l == 0) ? objOrig : objP[(l + 1) & 1];
    const __bf16* pin  = (l == 0) ? pred0   : predP[(l + 1) & 1];
    __bf16* oout = objP[l & 1];
    __bf16* pout = predP[l & 1];
    const int gi = 5 + l * 8;
    const float* b1a = (const float*)d_in[gi + 0];
    const float* b1b = (const float*)d_in[gi + 2];
    const float* b2a = (const float*)d_in[gi + 4];
    const float* b2b = (const float*)d_in[gi + 6];

    { GArgs g{}; g.M = T_EDGES; g.K = 3 * din; g.N = HID; g.Wt = wtG[l][0]; g.bias = b1a;
      g.objIn = oin; g.predIn = pin; g.din = din; g.sArr = sArr; g.oArr = oArr; g.Cbf = h1;
      gemm_wmma<A_GCONV, E_BF16><<<ggrid(g.M, g.N), 256, 0, stream>>>(g); }

    { long long n = (long long)O_NODES * HID;
      k_zero<<<(unsigned)((n + 255) / 256), 256, 0, stream>>>(pooled, n); }

    { GArgs g{}; g.M = T_EDGES; g.K = HID; g.N = 2 * HID + DD; g.Wt = wtG[l][1]; g.bias = b1b;
      g.Abf = h1; g.lda = HID; g.sArr = sArr; g.oArr = oArr; g.pooled = pooled; g.predOut = pout;
      gemm_wmma<A_BF16, E_SCATTER><<<ggrid(g.M, g.N), 256, 0, stream>>>(g); }

    { long long n = (long long)O_NODES * HID;
      k_divide<<<(unsigned)((n + 255) / 256), 256, 0, stream>>>(pooled, counts, n); }

    { GArgs g{}; g.M = O_NODES; g.K = HID; g.N = HID; g.Wt = wtG[l][2]; g.bias = b2a;
      g.Af32 = pooled; g.lda = HID; g.Cbf = h2;
      gemm_wmma<A_F32, E_BF16><<<ggrid(g.M, g.N), 256, 0, stream>>>(g); }

    { GArgs g{}; g.M = O_NODES; g.K = HID; g.N = DD; g.Wt = wtG[l][3]; g.bias = b2b;
      g.Abf = h2; g.lda = HID; g.Cbf = oout;
      gemm_wmma<A_BF16, E_BF16><<<ggrid(g.M, g.N), 256, 0, stream>>>(g); }
  }
  const __bf16* objFinal = objP[0];  // layer 4 output (l&1 == 0)

  // box head -> d_out[0 : 100000*4] (f32) + bf16 copy for rel gather
  { GArgs g{}; g.M = O_NODES; g.K = DD; g.N = HID; g.Wt = wtBox0; g.bias = box_b0;
    g.Abf = objFinal; g.lda = DD; g.Cbf = h2;
    gemm_wmma<A_BF16, E_BF16><<<ggrid(g.M, g.N), 256, 0, stream>>>(g); }
  { GArgs g{}; g.M = O_NODES; g.K = HID; g.N = 4; g.Wt = wtBox1; g.bias = box_b1;
    g.Abf = h2; g.lda = HID; g.Cf32 = (float*)d_out; g.Cbf2 = boxesBf;
    gemm_wmma<A_BF16, E_F32><<<ggrid(g.M, g.N), 256, 0, stream>>>(g); }

  // rel head -> d_out[100000*4 : +150000*46] (f32)
  { GArgs g{}; g.M = T_EDGES; g.K = 136; g.N = HID; g.Wt = wtRel0; g.bias = rel_b0;
    g.boxes = boxesBf; g.orig = objOrig; g.sArr = sArr; g.oArr = oArr; g.Cbf = h1;
    gemm_wmma<A_REL, E_BF16><<<ggrid(g.M, g.N), 256, 0, stream>>>(g); }
  { GArgs g{}; g.M = T_EDGES; g.K = HID; g.N = 46; g.Wt = wtRel1; g.bias = rel_b1;
    g.Abf = h1; g.lda = HID; g.Cf32 = (float*)d_out + (size_t)O_NODES * 4; g.Cbf2 = nullptr;
    gemm_wmma<A_BF16, E_F32><<<ggrid(g.M, g.N), 256, 0, stream>>>(g); }
}
